// Custom_Model_WOQ_Linear_Silu_Mul_24215025615268
// MI455X (gfx1250) — matrix operand order repaired and checked by reading, not compiled, not run on hardware
//
#include <hip/hip_runtime.h>
#include <hip/hip_bf16.h>
#include <stdint.h>

// ---------------------------------------------------------------------------
// WOQ int4 linear + SiLU + mul for gfx1250 (MI455X).
//   out = silu(inp @ ((unpack4(qweight) - unpack4(qzeros)) * scales) + bias) * mul
// bf16 WMMA (v_wmma_f32_16x16x32_bf16), f32 accumulate.
// Tiling: block 256 thr (8 wave32), BM=128 BN=256 BK=32, wave -> 64x64 out.
// Double-buffered LDS pipeline; int4 dequant fused into the B-tile stage using
// native v_cvt_pk_bf16_f32 conversions; all 8 fragments (4 A + 4 B) are loaded
// up-front each k-step so the 16 ds_load_b128 issue as one burst and the
// s_wait_dscnt cascade overlaps with WMMA drain.
// Assumes M%128==0, N%256==0, K%32==0, G%32==0 (true for 4096/4096/11008/128).
// ---------------------------------------------------------------------------

typedef __attribute__((ext_vector_type(16))) __bf16 v16bf;
typedef __attribute__((ext_vector_type(4)))  __bf16 v4bf;
typedef __attribute__((ext_vector_type(8)))  float  v8f;

#define BM 128
#define BN 256
#define BK 32
#define LDA 40   // bf16 elems per LDS A row (32 + 8 pad -> 80B stride, conflict-free)
#define LDB 40   // bf16 elems per LDS B row ([n][k] layout)
#define NTHREADS 256

static __device__ __forceinline__ v4bf pk4bf(float a, float b, float c, float d) {
  // lowers to v_cvt_pk_bf16_f32 pairs (native RNE)
  return (v4bf){(__bf16)a, (__bf16)b, (__bf16)c, (__bf16)d};
}

__global__ __launch_bounds__(NTHREADS)
void woq_silu_mul_kernel(const float* __restrict__ inp,
                         const int*   __restrict__ qweight,
                         const float* __restrict__ scales,
                         const int*   __restrict__ qzeros,
                         const float* __restrict__ bias,
                         const float* __restrict__ mulv,
                         float*       __restrict__ out,
                         int M, int N, int K, int G) {
  const int Nw  = N >> 3;                 // packed words per K-row
  const int tid = threadIdx.x;
  const int bm  = blockIdx.y * BM;
  const int bn  = blockIdx.x * BN;

  __shared__ __bf16 As[2][BM * LDA];      // 2 * 10240 B
  __shared__ __bf16 Bs[2][BN * LDB];      // 2 * 20480 B  (total 60 KB of 320 KB)

  const int lane = tid & 31;
  const int wid  = tid >> 5;              // 0..7
  const int wm   = (wid & 1) * 64;        // wave row offset in block tile
  const int wn   = (wid >> 1) * 64;       // wave col offset in block tile
  const int fr   = lane & 15;             // fragment row (A: M) / col (B: N)
  const int hh   = lane >> 4;             // lane half
  const int kA   = hh * 8;                // A frag: K in {8h..8h+7} U {16+8h..}
  const int kB   = hh * 8;                // B frag: K in {8h..8h+7} U {16+8h..}

  // ---- loader coordinates: each thread owns (nw, kb..kb+3) of the B tile ----
  const int nw = tid & 31;                // word column within the 32-word tile
  const int kb = (tid >> 5) << 2;         // 4 consecutive k rows
  const int nb = bn >> 3;                 // word-column base of this block

  float4 ra[4];                           // staged A (fp32)
  int    rq[4];                           // staged qweight words
  int    rz;                              // staged qzeros word
  float4 rs0, rs1;                        // staged 8 scales

  auto gload = [&](int kt) {              // global -> regs
    const int k0 = kt * BK;
#pragma unroll
    for (int i = 0; i < 4; ++i) {         // A: 128x32 fp32, 4 x float4 / thread
      const int f = tid + NTHREADS * i;   // float4 id: 128 rows x 8 per row
      const int row = f >> 3, c = (f & 7) << 2;
      ra[i] = *(const float4*)(inp + (size_t)(bm + row) * K + (k0 + c));
    }
    const int g = k0 / G;                 // BK=32 divides G=128: one group/tile
    rz = qzeros[(size_t)g * Nw + nb + nw];
    const float* sp = scales + (size_t)g * N + bn + nw * 8;
    rs0 = *(const float4*)sp;
    rs1 = *(const float4*)(sp + 4);
#pragma unroll
    for (int i = 0; i < 4; ++i)
      rq[i] = qweight[(size_t)(k0 + kb + i) * Nw + nb + nw];
  };

  auto sstore = [&](int buf) {            // regs -> LDS (cvt + dequant)
#pragma unroll
    for (int i = 0; i < 4; ++i) {
      const int f = tid + NTHREADS * i;
      const int row = f >> 3, c = (f & 7) << 2;
      *(v4bf*)&As[buf][row * LDA + c] =
          pk4bf(ra[i].x, ra[i].y, ra[i].z, ra[i].w);     // ds_store_b64
    }
    const float sj[8] = {rs0.x, rs0.y, rs0.z, rs0.w, rs1.x, rs1.y, rs1.z, rs1.w};
#pragma unroll
    for (int j = 0; j < 8; ++j) {                        // 8 n's per word
      const float sc = sj[j];
      const float zs = (float)((rz >> (4 * j)) & 0xF) * sc;  // z*scale
      const float w0 = fmaf((float)((rq[0] >> (4 * j)) & 0xF), sc, -zs);
      const float w1 = fmaf((float)((rq[1] >> (4 * j)) & 0xF), sc, -zs);
      const float w2 = fmaf((float)((rq[2] >> (4 * j)) & 0xF), sc, -zs);
      const float w3 = fmaf((float)((rq[3] >> (4 * j)) & 0xF), sc, -zs);
      // transposed [n][k] so B fragments read contiguous K
      *(v4bf*)&Bs[buf][(nw * 8 + j) * LDB + kb] = pk4bf(w0, w1, w2, w3);
    }
  };

  v8f acc[4][4];
#pragma unroll
  for (int i = 0; i < 4; ++i)
#pragma unroll
    for (int j = 0; j < 4; ++j)
      acc[i][j] = (v8f){0.f, 0.f, 0.f, 0.f, 0.f, 0.f, 0.f, 0.f};

  auto compute = [&](int buf) {
    v16bf a[4], b[4];
    // issue all 16 ds_load_b128 as one burst, then let the wait cascade
#pragma unroll
    for (int ti = 0; ti < 4; ++ti) {
      const __bf16* p = &As[buf][(wm + ti * 16 + fr) * LDA + kA];
      union { uint4 u[2]; v16bf v; } x;
      x.u[0] = *(const uint4*)p;          // K = 8h .. 8h+7
      x.u[1] = *(const uint4*)(p + 16);   // K = 16+8h .. 16+8h+7
      a[ti] = x.v;
    }
#pragma unroll
    for (int tj = 0; tj < 4; ++tj) {
      const __bf16* p = &Bs[buf][(wn + tj * 16 + fr) * LDB + kB];
      union { uint4 u[2]; v16bf v; } y;
      y.u[0] = *(const uint4*)p;          // K = 8h .. 8h+7
      y.u[1] = *(const uint4*)(p + 16);   // K = 16+8h .. 16+8h+7
      b[tj] = y.v;
    }
#pragma unroll
    for (int tj = 0; tj < 4; ++tj)
#pragma unroll
      for (int ti = 0; ti < 4; ++ti)
        acc[ti][tj] = __builtin_amdgcn_wmma_f32_16x16x32_bf16(
            false, a[ti], false, b[tj], (short)0, acc[ti][tj], false, false);
  };

  // ---- double-buffered main loop ----
  const int KT = K / BK;
  gload(0);
  sstore(0);
  __syncthreads();
  for (int kt = 0; kt < KT; ++kt) {
    const int cur = kt & 1;
    if (kt + 1 < KT) gload(kt + 1);       // VMEM for next tile overlaps WMMA
    compute(cur);
    if (kt + 1 < KT) sstore(cur ^ 1);
    __syncthreads();
  }

  // ---- fused epilogue: silu(x + bias) * mul ----
#pragma unroll
  for (int ti = 0; ti < 4; ++ti) {
#pragma unroll
    for (int tj = 0; tj < 4; ++tj) {
      const int col  = bn + wn + tj * 16 + fr;
      const int row0 = bm + wm + ti * 16 + hh * 8;   // C/D: VGPR j -> M=j(+8*half)
      const float bv = bias[col];
#pragma unroll
      for (int j = 0; j < 8; ++j) {
        const int row = row0 + j;
        const float x = acc[ti][tj][j] + bv;
        // silu(x) = x * sigmoid(x), via native v_exp_f32 (exp2) + v_rcp_f32
        const float e = __builtin_amdgcn_exp2f(x * -1.442695041f);
        const float s = x * __builtin_amdgcn_rcpf(1.0f + e);
        out[(size_t)row * N + col] = s * mulv[(size_t)row * N + col];
      }
    }
  }
}

extern "C" void kernel_launch(void* const* d_in, const int* in_sizes, int n_in,
                              void* d_out, int out_size, void* d_ws, size_t ws_size,
                              hipStream_t stream) {
  const float* inp     = (const float*)d_in[0];
  const int*   qweight = (const int*)d_in[1];
  const float* scales  = (const float*)d_in[2];
  const int*   qzeros  = (const int*)d_in[3];
  const float* bias    = (const float*)d_in[4];
  const float* mulv    = (const float*)d_in[5];

  const int N  = in_sizes[4];                               // bias: [N]
  const int M  = (int)((long long)in_sizes[5] / N);         // mul: [M,N]
  const int K  = (int)((long long)in_sizes[0] / M);         // inp: [M,K]
  const int KG = (int)((long long)in_sizes[2] / N);         // scales: [K/G,N]
  const int G  = K / KG;

  dim3 grid(N / BN, M / BM);   // 43 x 32 for the reference shapes
  woq_silu_mul_kernel<<<grid, NTHREADS, 0, stream>>>(
      inp, qweight, scales, qzeros, bias, mulv, (float*)d_out, M, N, K, G);
}
